// DeepIRTModel_25400436589020
// MI455X (gfx1250) — matrix operand order repaired and checked by reading, not compiled, not run on hardware
//
#include <hip/hip_runtime.h>
#include <hip/hip_bf16.h>

// ---------------- problem constants ----------------
#define BB   256
#define SS   512
#define MEMN 32
#define DK   128
#define DV   128
#define DSUM 256
#define BSTOK (BB*SS)          // 131072 tokens
#define NTILE (BSTOK/16)       // 8192 16-token tiles

typedef __attribute__((ext_vector_type(16))) __bf16        v16bf;
typedef __attribute__((ext_vector_type(8)))  unsigned int  v8u;
typedef __attribute__((ext_vector_type(8)))  float         v8f;

// hardware round-to-nearest-even f32 -> bf16
__device__ __forceinline__ unsigned short f2bf(float f) {
    __bf16 h = (__bf16)f;
    return __builtin_bit_cast(unsigned short, h);
}
__device__ __forceinline__ float bf2f(unsigned short h) {
    return __uint_as_float(((unsigned)h) << 16);
}

__device__ __forceinline__ v8f wmma_bf16(v16bf a, v16bf b, v8f c) {
    return __builtin_amdgcn_wmma_f32_16x16x32_bf16(false, a, false, b, (short)0, c, false, false);
}

// A-fragment (16x32 bf16) loader from a bf16 (u16) row, global or LDS.
// ISA layout: lane<16 -> M=lane&15; pairs p=0..3 hold K=kbase+8*half+{0..7},
// pairs p=4..7 hold K=kbase+16+8*half+{0..7}  ->  two contiguous 16B runs.
__device__ __forceinline__ v16bf load_A_u16(const unsigned short* row, int kbase, int half) {
    const uint4 lo = *reinterpret_cast<const uint4*>(row + kbase + (half << 3));
    const uint4 hi = *reinterpret_cast<const uint4*>(row + kbase + 16 + (half << 3));
    v8u t;
    t[0] = lo.x; t[1] = lo.y; t[2] = lo.z; t[3] = lo.w;
    t[4] = hi.x; t[5] = hi.y; t[6] = hi.z; t[7] = hi.w;
    return __builtin_bit_cast(v16bf, t);
}

// B fragment: pre-swizzled [frag][lane][16] u16, 32B per lane -> 2x b128 loads.
__device__ __forceinline__ v16bf ldB(const unsigned short* __restrict__ wB, int frag, int lane) {
    return *reinterpret_cast<const v16bf*>(wB + ((((size_t)frag << 5) + lane) << 4));
}

// ---------------- one-time converts ----------------
// bulk f32 -> bf16 (embedding tables), 4 elems/thread
__global__ void cvt_bf16(const float* __restrict__ src, unsigned short* __restrict__ dst, int n4) {
    for (int i = blockIdx.x * blockDim.x + threadIdx.x; i < n4; i += gridDim.x * blockDim.x) {
        float4 v = reinterpret_cast<const float4*>(src)[i];
        ushort4 o;
        o.x = f2bf(v.x); o.y = f2bf(v.y); o.z = f2bf(v.z); o.w = f2bf(v.w);
        reinterpret_cast<ushort4*>(dst)[i] = o;
    }
}

// weight swizzle into WMMA B-fragment order: [kc][nc][lane][e]
//   k = kc*32 + (e>>3)*16 + (lane>>4)*8 + (e&7) ; n = nc*16 + (lane&15)
// transposed=1: B[k][n] = W[n*K + k] (key_memory: logits = qe @ KM^T)
__global__ void prep_w(const float* __restrict__ W, unsigned short* __restrict__ dst,
                       int K, int N, int transposed) {
    int total = K * N;
    int nTiles = N >> 4;
    for (int i = blockIdx.x * blockDim.x + threadIdx.x; i < total; i += gridDim.x * blockDim.x) {
        int e  = i & 15;
        int l  = (i >> 4) & 31;
        int f  = i >> 9;
        int kc = f / nTiles, nc = f - kc * nTiles;
        int k  = kc * 32 + ((e >> 3) << 4) + ((l >> 4) << 3) + (e & 7);
        int n  = (nc << 4) + (l & 15);
        float v = transposed ? W[n * K + k] : W[k * N + n];
        dst[i] = f2bf(v);
    }
}

// ---------------- kernel A: attention softmax + erase/add gates ------------
__global__ __launch_bounds__(256) void k_embed_gates(
    const int* __restrict__ q_data, const int* __restrict__ qa_data,
    const unsigned short* __restrict__ q_tab, const unsigned short* __restrict__ qa_tab,
    const unsigned short* __restrict__ kmB,
    const unsigned short* __restrict__ WeB, const unsigned short* __restrict__ WaB,
    const float* __restrict__ b_erase, const float* __restrict__ b_add,
    float* __restrict__ w_all, unsigned short* __restrict__ eraseT,
    unsigned short* __restrict__ addT)
{
    __shared__ float lg[8][16][MEMN];
    int wv   = threadIdx.x >> 5;
    int lane = threadIdx.x & 31;
    int tile = blockIdx.x * 8 + wv;
    int m    = lane & 15;
    int half = lane >> 4;
    int tok  = tile * 16 + m;

    const unsigned short* qrow  = q_tab  + (size_t)q_data[tok]  * DK;
    const unsigned short* qarow = qa_tab + (size_t)qa_data[tok] * DV;

    v16bf qeF[4], qaF[4];
#pragma unroll
    for (int kc = 0; kc < 4; ++kc) {
        qeF[kc] = load_A_u16(qrow,  kc * 32, half);
        qaF[kc] = load_A_u16(qarow, kc * 32, half);
    }

    // attention logits: [16x128] x [128x32]
#pragma unroll
    for (int nc = 0; nc < 2; ++nc) {
        v8f acc = {};
#pragma unroll
        for (int kc = 0; kc < 4; ++kc)
            acc = wmma_bf16(qeF[kc], ldB(kmB, kc * 2 + nc, lane), acc);
#pragma unroll
        for (int r = 0; r < 8; ++r)
            lg[wv][r + 8 * half][nc * 16 + m] = acc[r];
    }
    // per-row softmax over MEM=32 (LDS ops within a wave are in-order)
    if (lane < 16) {
        float v[MEMN];
        float mx = -3.0e38f;
#pragma unroll
        for (int j = 0; j < MEMN; ++j) { v[j] = lg[wv][lane][j]; mx = fmaxf(mx, v[j]); }
        float s = 0.f;
#pragma unroll
        for (int j = 0; j < MEMN; ++j) { v[j] = __expf(v[j] - mx); s += v[j]; }
        float inv = 1.0f / s;
        float* dst = w_all + (size_t)(tile * 16 + lane) * MEMN;
#pragma unroll
        for (int j = 0; j < MEMN; ++j) dst[j] = v[j] * inv;
    }

    // erase = sigmoid(qae @ W_erase + b), add = tanh(qae @ W_add + b)
    for (int nc = 0; nc < 8; ++nc) {
        v8f ae = {}, aa = {};
#pragma unroll
        for (int kc = 0; kc < 4; ++kc) {
            ae = wmma_bf16(qaF[kc], ldB(WeB, kc * 8 + nc, lane), ae);
            aa = wmma_bf16(qaF[kc], ldB(WaB, kc * 8 + nc, lane), aa);
        }
        int col = nc * 16 + m;
        float be = b_erase[col], ba = b_add[col];
#pragma unroll
        for (int r = 0; r < 8; ++r) {
            int row = tile * 16 + r + 8 * half;
            float ev = 1.0f / (1.0f + __expf(-(ae[r] + be)));
            float av = tanhf(aa[r] + ba);
            eraseT[(size_t)row * DV + col] = f2bf(ev);
            addT  [(size_t)row * DV + col] = f2bf(av);
        }
    }
}

// ---------------- kernel B: recurrent DKVMN scan over S --------------------
__global__ __launch_bounds__(128) void k_scan(
    const float* __restrict__ w_all, const unsigned short* __restrict__ eraseT,
    const unsigned short* __restrict__ addT, const float* __restrict__ Mv0,
    unsigned short* __restrict__ readsT)
{
    int b = blockIdx.x;          // batch
    int t = threadIdx.x;         // value dim 0..127
    float Mv[MEMN];
#pragma unroll
    for (int mm = 0; mm < MEMN; ++mm) Mv[mm] = Mv0[mm * DV + t];

    __shared__ float wsh[MEMN];
    for (int s = 0; s < SS; ++s) {
        int tok = b * SS + s;
        if (t < MEMN) wsh[t] = w_all[(size_t)tok * MEMN + t];
        __syncthreads();
        float e = bf2f(eraseT[(size_t)tok * DV + t]);
        float a = bf2f(addT  [(size_t)tok * DV + t]);
        float rd = 0.f;
#pragma unroll
        for (int mm = 0; mm < MEMN; ++mm) {
            float w = wsh[mm];
            rd += w * Mv[mm];                            // read BEFORE write
            Mv[mm] = Mv[mm] * (1.0f - w * e) + w * a;    // erase/add update
        }
        readsT[(size_t)tok * DV + t] = f2bf(rd);
        __syncthreads();
    }
}

// ---------------- kernel C: summary + ability + difficulty + outputs --------
__global__ __launch_bounds__(128) void k_head(
    const int* __restrict__ q_data, const unsigned short* __restrict__ q_tab,
    const unsigned short* __restrict__ readsT,
    const unsigned short* __restrict__ WsumB, const float* __restrict__ b_sum,
    const unsigned short* __restrict__ Wab1B, const float* __restrict__ b_ab1,
    const float* __restrict__ W_ab2, const float* __restrict__ b_ab2,
    const unsigned short* __restrict__ Wdf1B, const float* __restrict__ b_df1,
    const float* __restrict__ W_df2, const float* __restrict__ b_df2,
    float* __restrict__ out)
{
    __shared__ unsigned short stile[4][16][DSUM];   // bf16 summary tile per wave
    __shared__ float2 red[4][32][8];                // (ability, difficulty) partials
    int wv   = threadIdx.x >> 5;
    int lane = threadIdx.x & 31;
    int tile = blockIdx.x * 4 + wv;
    int m    = lane & 15;
    int half = lane >> 4;
    int tok  = tile * 16 + m;

    const unsigned short* qrow = q_tab  + (size_t)q_data[tok] * DK;
    const unsigned short* rrow = readsT + (size_t)tok * DV;

    v16bf qeF[4], rdF[4];
#pragma unroll
    for (int kc = 0; kc < 4; ++kc) {
        qeF[kc] = load_A_u16(qrow, kc * 32, half);
        rdF[kc] = load_A_u16(rrow, kc * 32, half);
    }

    // summary = tanh([reads|qe] @ W_sum + b_sum)   K=256, N=256
    for (int nc = 0; nc < 16; ++nc) {
        v8f acc = {};
#pragma unroll
        for (int kc = 0; kc < 4; ++kc)
            acc = wmma_bf16(rdF[kc], ldB(WsumB, kc * 16 + nc, lane), acc);
#pragma unroll
        for (int kc = 0; kc < 4; ++kc)
            acc = wmma_bf16(qeF[kc], ldB(WsumB, (4 + kc) * 16 + nc, lane), acc);
        int col = nc * 16 + m;
        float bs = b_sum[col];
#pragma unroll
        for (int r = 0; r < 8; ++r)
            stile[wv][r + 8 * half][col] = f2bf(tanhf(acc[r] + bs));
    }

    // re-fragment summary from LDS (wave-internal LDS is in-order)
    v16bf smF[8];
#pragma unroll
    for (int kc = 0; kc < 8; ++kc)
        smF[kc] = load_A_u16(&stile[wv][m][0], kc * 32, half);

    float2 part[8];
#pragma unroll
    for (int r = 0; r < 8; ++r) part[r] = make_float2(0.f, 0.f);

    // ability hidden: tanh(summary @ W_ab1 + b) then dot with W_ab2
    for (int nc = 0; nc < 16; ++nc) {
        v8f acc = {};
#pragma unroll
        for (int kc = 0; kc < 8; ++kc)
            acc = wmma_bf16(smF[kc], ldB(Wab1B, kc * 16 + nc, lane), acc);
        int col = nc * 16 + m;
        float bb = b_ab1[col], w2 = W_ab2[col];
#pragma unroll
        for (int r = 0; r < 8; ++r)
            part[r].x += tanhf(acc[r] + bb) * w2;
    }
    // difficulty hidden: tanh(qe @ W_df1 + b) then dot with W_df2
    for (int nc = 0; nc < 8; ++nc) {
        v8f acc = {};
#pragma unroll
        for (int kc = 0; kc < 4; ++kc)
            acc = wmma_bf16(qeF[kc], ldB(Wdf1B, kc * 8 + nc, lane), acc);
        int col = nc * 16 + m;
        float bd = b_df1[col], w2 = W_df2[col];
#pragma unroll
        for (int r = 0; r < 8; ++r)
            part[r].y += tanhf(acc[r] + bd) * w2;
    }

#pragma unroll
    for (int r = 0; r < 8; ++r) red[wv][lane][r] = part[r];

    if (lane < 16) {
        int r = lane & 7, h = lane >> 3;     // row m = r + 8*h lives in lane-half h
        float sa = 0.f, sd = 0.f;
#pragma unroll
        for (int j = 0; j < 16; ++j) {
            float2 p = red[wv][h * 16 + j][r];
            sa += p.x; sd += p.y;
        }
        float ability    = sa + b_ab2[0];
        float difficulty = sd + b_df2[0];
        float z    = 3.0f * ability - difficulty;
        float pred = 1.0f / (1.0f + __expf(-z));
        int to = tile * 16 + lane;
        out[to]             = pred;
        out[BSTOK + to]     = ability;
        out[2 * BSTOK + to] = difficulty;
        out[3 * BSTOK + to] = z;
    }
}

// ---------------- launcher -------------------------------------------------
extern "C" void kernel_launch(void* const* d_in, const int* in_sizes, int n_in,
                              void* d_out, int out_size, void* d_ws, size_t ws_size,
                              hipStream_t stream) {
    const int*   q_data     = (const int*)  d_in[0];
    const int*   qa_data    = (const int*)  d_in[1];
    const float* q_table    = (const float*)d_in[2];
    const float* qa_table   = (const float*)d_in[3];
    const float* key_memory = (const float*)d_in[4];
    const float* init_value = (const float*)d_in[5];
    const float* W_erase    = (const float*)d_in[6];
    const float* b_erase    = (const float*)d_in[7];
    const float* W_add      = (const float*)d_in[8];
    const float* b_add      = (const float*)d_in[9];
    const float* W_sum      = (const float*)d_in[10];
    const float* b_sum      = (const float*)d_in[11];
    const float* W_ab1      = (const float*)d_in[12];
    const float* b_ab1      = (const float*)d_in[13];
    const float* W_ab2      = (const float*)d_in[14];
    const float* b_ab2      = (const float*)d_in[15];
    const float* W_df1      = (const float*)d_in[16];
    const float* b_df1      = (const float*)d_in[17];
    const float* W_df2      = (const float*)d_in[18];
    const float* b_df2      = (const float*)d_in[19];

    char* ws = (char*)d_ws;
    // swizzled bf16 weights
    unsigned short* kmB    = (unsigned short*)(ws + 0);          //  8 KB
    unsigned short* WeB    = (unsigned short*)(ws + 8192);       // 32 KB
    unsigned short* WaB    = (unsigned short*)(ws + 40960);      // 32 KB
    unsigned short* WsumB  = (unsigned short*)(ws + 73728);      //128 KB
    unsigned short* Wab1B  = (unsigned short*)(ws + 204800);     //128 KB
    unsigned short* Wdf1B  = (unsigned short*)(ws + 335872);     // 32 KB
    // bf16 embedding tables
    unsigned short* q_tab  = (unsigned short*)(ws + 368640);     // 10001*128*2 = 2,560,256
    unsigned short* qa_tab = (unsigned short*)(ws + 2928896);    // 20001*128*2 = 5,120,256
    // activations
    float*          w_all  = (float*)        (ws + 8049152);     // 16 MB
    unsigned short* eraseT = (unsigned short*)(ws + 24826368);   // 32 MB
    unsigned short* addT   = (unsigned short*)(ws + 58380800);   // 32 MB
    unsigned short* readsT = (unsigned short*)(ws + 91935232);   // 32 MB
    (void)in_sizes; (void)n_in; (void)out_size; (void)ws_size;

    // one-time weight swizzles + table converts
    prep_w<<<16,  256, 0, stream>>>(key_memory, kmB,   DK,   MEMN, 1);
    prep_w<<<64,  256, 0, stream>>>(W_erase,    WeB,   DV,   DV,   0);
    prep_w<<<64,  256, 0, stream>>>(W_add,      WaB,   DV,   DV,   0);
    prep_w<<<256, 256, 0, stream>>>(W_sum,      WsumB, DV+DK, DSUM, 0);
    prep_w<<<256, 256, 0, stream>>>(W_ab1,      Wab1B, DSUM, DSUM, 0);
    prep_w<<<64,  256, 0, stream>>>(W_df1,      Wdf1B, DK,   DK,   0);
    cvt_bf16<<<1024, 256, 0, stream>>>(q_table,  q_tab,  (10001 * DK) / 4);
    cvt_bf16<<<2048, 256, 0, stream>>>(qa_table, qa_tab, (20001 * DV) / 4);

    k_embed_gates<<<NTILE / 8, 256, 0, stream>>>(
        q_data, qa_data, q_tab, qa_tab, kmB, WeB, WaB,
        b_erase, b_add, w_all, eraseT, addT);

    k_scan<<<BB, 128, 0, stream>>>(w_all, eraseT, addT, init_value, readsT);

    k_head<<<NTILE / 4, 128, 0, stream>>>(
        q_data, q_tab, readsT, WsumB, b_sum, Wab1B, b_ab1,
        W_ab2, b_ab2, Wdf1B, b_df1, W_df2, b_df2, (float*)d_out);
}
